// ExLRestSelfAtten_49125835931701
// MI455X (gfx1250) — matrix operand order repaired and checked by reading, not compiled, not run on hardware
//
#include <hip/hip_runtime.h>

// ---------- CDNA5 WMMA types ----------
typedef __attribute__((ext_vector_type(16))) __bf16 v16bf;
typedef __attribute__((ext_vector_type(8)))  float  v8f;

struct U4 { unsigned int x, y, z, w; };   // POD 16-byte chunk (8 bf16)

union Frag {
  v16bf v;
  U4 q[2];
};

__device__ __forceinline__ unsigned short f2bfu(float f) {
  unsigned int x = __float_as_uint(f);
  x += 0x7FFFu + ((x >> 16) & 1u);       // round-to-nearest-even
  return (unsigned short)(x >> 16);
}
__device__ __forceinline__ float bf2f(unsigned short u) {
  return __uint_as_float(((unsigned int)u) << 16);
}

__device__ __forceinline__ v8f wmma_bf16(Frag a, Frag b, v8f c) {
  return __builtin_amdgcn_wmma_f32_16x16x32_bf16(
      /*neg_a=*/false, a.v, /*neg_b=*/false, b.v,
      /*c_mod=*/(short)0, c, /*reuse_a=*/false, /*reuse_b=*/false);
}

// ---------- helpers: convert / pack ----------
__global__ __launch_bounds__(256) void k_f32_to_bf16(const float* __restrict__ src,
                                                     unsigned short* __restrict__ dst, int n) {
  int i = blockIdx.x * 256 + threadIdx.x;
  if (i < n) dst[i] = f2bfu(src[i]);
}

// Pack fp32 weight W[K][N] into B-fragment layout:
// dst[((nTile*KT + kTile)*32 + lane)*16 + j] = bf16(W[kTile*32 + 16*(lane/16) + j][nTile*16 + lane%16])
__global__ __launch_bounds__(256) void k_pack_w(const float* __restrict__ W,
                                                unsigned short* __restrict__ dst,
                                                int K, int N) {
  int idx = blockIdx.x * 256 + threadIdx.x;
  if (idx >= K * N) return;
  int j    = idx & 15;
  int lane = (idx >> 4) & 31;
  int t    = idx >> 9;            // nTile*KT + kTile
  int KT   = K >> 5;
  int kTile = t % KT;
  int nTile = t / KT;
  int k = kTile * 32 + ((lane >> 4) << 4) + j;
  int n = nTile * 16 + (lane & 15);
  dst[idx] = f2bfu(W[k * N + n]);
}

// ---------- WMMA GEMM: C[M,N] = act(A[M,K] * Wpacked + bias) ----------
// A: bf16 row-major. One wave per block; each wave: 64x64 macro-tile
// (4 M-tiles x 4 N-tiles -> 16 WMMAs per 32-wide K-chunk; B frags reused 4x).
__global__ __launch_bounds__(32) void k_gemm(const unsigned short* __restrict__ A,
                                             const unsigned short* __restrict__ Bp,
                                             const float* __restrict__ bias,
                                             float* __restrict__ Cf,
                                             unsigned short* __restrict__ Cbf,
                                             int K, int N, int relu) {
  int lane  = threadIdx.x;
  int mbase = blockIdx.x << 6;           // 64 rows per wave
  int nt0   = blockIdx.y << 2;           // 4 N-tiles per wave
  int KT    = K >> 5;
  const unsigned short* arow0 =
      A + (size_t)(mbase + (lane & 15)) * K + ((lane >> 4) << 4);
  const size_t tstride = (size_t)KT * 512;   // ushorts per N-tile block
  const unsigned short* bbase = Bp + (size_t)nt0 * tstride + ((size_t)lane << 4);

  v8f zero = {};
  v8f acc[16];
#pragma unroll
  for (int i = 0; i < 16; ++i) acc[i] = zero;

  for (int kc = 0; kc < KT; ++kc) {
    Frag a[4], b[4];
#pragma unroll
    for (int m = 0; m < 4; ++m) {
      const U4* ap = (const U4*)(arow0 + (size_t)m * 16 * K + (kc << 5));
      a[m].q[0] = ap[0];            // K = kc*32 + 16*(lane/16) + 0..7
      a[m].q[1] = ap[1];            // K = kc*32 + 16*(lane/16) + 8..15
    }
#pragma unroll
    for (int n = 0; n < 4; ++n) {
      const U4* bp = (const U4*)(bbase + (size_t)n * tstride + (size_t)kc * 512);
      b[n].q[0] = bp[0];
      b[n].q[1] = bp[1];
    }
#pragma unroll
    for (int m = 0; m < 4; ++m)
#pragma unroll
      for (int n = 0; n < 4; ++n)
        acc[m * 4 + n] = wmma_bf16(a[m], b[n], acc[m * 4 + n]);
  }

  // epilogue: lane holds col N=lane%16, rows m = rb..rb+7 within each tile
  int ncl = lane & 15;
  int rb  = (lane >> 4) << 3;
#pragma unroll
  for (int m = 0; m < 4; ++m) {
#pragma unroll
    for (int n = 0; n < 4; ++n) {
      int col = ((nt0 + n) << 4) + ncl;
      float bv = bias ? bias[col] : 0.0f;
      v8f av = acc[m * 4 + n];
#pragma unroll
      for (int r = 0; r < 8; ++r) {
        float v = av[r] + bv;
        if (relu) v = fmaxf(v, 0.0f);
        size_t off = (size_t)(mbase + m * 16 + rb + r) * N + col;
        if (Cbf) Cbf[off] = f2bfu(v);
        else     Cf[off]  = v;
      }
    }
  }
}

// ---------- banded Gram matrix ----------
// G[b,i,j] = qf[b,i] . kf[b,j] / sqrt(512), computed only for the band
// |i-j| <= 32 (tile granularity: jt in [it-2, it+2]). Each banded pair is
// computed exactly once instead of ~8x in the naive per-position form.
__global__ __launch_bounds__(32) void k_scores(const unsigned short* __restrict__ qf,
                                               const unsigned short* __restrict__ kf,
                                               float* __restrict__ G) {
  int b  = blockIdx.x;                 // 64
  int it = blockIdx.y;                 // 7 (112 rows padded from 100)
  int jt = it - 2 + blockIdx.z;        // 5 candidate j-tiles
  if (jt < 0 || jt > 6) return;
  int lane = threadIdx.x;

  int i = it * 16 + (lane & 15);
  int j = jt * 16 + (lane & 15);
  bool va = (i < 100);
  bool vb = (j < 100);
  const unsigned short* aptr =
      qf + (size_t)(b * 100 + (va ? i : 0)) * 512 + ((lane >> 4) << 4);
  const unsigned short* bptr =
      kf + (size_t)(b * 100 + (vb ? j : 0)) * 512 + ((lane >> 4) << 4);

  const U4 z4 = {0u, 0u, 0u, 0u};
  v8f acc = {};
  for (int kc = 0; kc < 16; ++kc) {
    Frag a, bb;
    const U4* ap = (const U4*)(aptr + (kc << 5));
    a.q[0] = va ? ap[0] : z4;
    a.q[1] = va ? ap[1] : z4;
    const U4* bp = (const U4*)(bptr + (kc << 5));
    bb.q[0] = vb ? bp[0] : z4;
    bb.q[1] = vb ? bp[1] : z4;
    acc = wmma_bf16(a, bb, acc);
  }

  const float scale = 0.04419417382415922f;  // 1/sqrt(512)
  int col = jt * 16 + (lane & 15);
  int rb  = it * 16 + ((lane >> 4) << 3);
#pragma unroll
  for (int r = 0; r < 8; ++r) {
    int row = rb + r;
    if (row < 100 && col < 100)
      G[((size_t)b * 100 + row) * 100 + col] = acc[r] * scale;
  }
}

// ---------- windowed attention from banded G ----------
// One block per (b,s): gather d[w][u] = G[b, s+16-w, s+16-u] (0 outside),
// softmax rows over u, write alpha, beta[u] = sum_w alpha[w][u],
// out[h] = sum_u beta[u] * v[s+16-u][h].
#define WIN 33
__global__ __launch_bounds__(256) void k_atten(const float* __restrict__ G,
                                               const unsigned short* __restrict__ vf,
                                               float* __restrict__ alphaOut,
                                               unsigned short* __restrict__ attnOut) {
  __shared__ float sS[WIN * 34];
  __shared__ float sBeta[WIN];
  int bs = blockIdx.x;
  int b = bs / 100, s = bs % 100;
  int tid = threadIdx.x;

  for (int e = tid; e < WIN * WIN; e += 256) {
    int w = e / WIN, u = e - w * WIN;
    int i = s + 16 - w;
    int j = s + 16 - u;
    float v = 0.0f;
    if (i >= 0 && i < 100 && j >= 0 && j < 100)
      v = G[((size_t)b * 100 + i) * 100 + j];
    sS[w * 34 + u] = v;
  }
  __syncthreads();

  // rowwise softmax over u; write alpha to global and back to LDS
  if (tid < WIN) {
    float m = -3.0e38f;
    for (int j = 0; j < WIN; ++j) m = fmaxf(m, sS[tid * 34 + j]);
    float sum = 0.f;
    for (int j = 0; j < WIN; ++j) sum += __expf(sS[tid * 34 + j] - m);
    float inv = 1.0f / sum;
    float* arow = alphaOut + (size_t)(bs * WIN + tid) * WIN;
    for (int j = 0; j < WIN; ++j) {
      float aij = __expf(sS[tid * 34 + j] - m) * inv;
      arow[j] = aij;
      sS[tid * 34 + j] = aij;
    }
  }
  __syncthreads();

  if (tid < WIN) {
    float bsum = 0.f;
    for (int ww = 0; ww < WIN; ++ww) bsum += sS[ww * 34 + tid];
    sBeta[tid] = bsum;
  }
  __syncthreads();

  for (int h = tid; h < 512; h += 256) {
    float o = 0.f;
    for (int j = 0; j < WIN; ++j) {
      int kr = s + 16 - j;
      if (kr >= 0 && kr < 100)
        o += sBeta[j] * bf2f(vf[(size_t)(b * 100 + kr) * 512 + h]);
    }
    attnOut[(size_t)bs * 512 + h] = f2bfu(o);
  }
}

// ---------- launcher ----------
extern "C" void kernel_launch(void* const* d_in, const int* in_sizes, int n_in,
                              void* d_out, int out_size, void* d_ws, size_t ws_size,
                              hipStream_t stream) {
  const float* x  = (const float*)d_in[0];
  const float* W0 = (const float*)d_in[1];
  const float* b0 = (const float*)d_in[2];
  const float* Wq = (const float*)d_in[3];
  const float* Wk = (const float*)d_in[4];
  const float* Wv = (const float*)d_in[5];
  const float* W1 = (const float*)d_in[6];
  const float* b1 = (const float*)d_in[7];

  float* y        = (float*)d_out;                 // 64*100*256
  float* alphaOut = y + 64 * 100 * 256;            // 6400*33*33

  char* ws = (char*)d_ws;
  size_t off = 0;
  auto alloc = [&](size_t bytes) {
    char* p = ws + off;
    off += (bytes + 255) & ~(size_t)255;
    return p;
  };
  unsigned short* xbf   = (unsigned short*)alloc(6400ull * 256 * 2);
  unsigned short* W0p   = (unsigned short*)alloc(256ull * 512 * 2);
  unsigned short* Wqp   = (unsigned short*)alloc(512ull * 512 * 2);
  unsigned short* Wkp   = (unsigned short*)alloc(512ull * 512 * 2);
  unsigned short* Wvp   = (unsigned short*)alloc(512ull * 512 * 2);
  unsigned short* W1p   = (unsigned short*)alloc(512ull * 256 * 2);
  unsigned short* h     = (unsigned short*)alloc(6400ull * 512 * 2);
  unsigned short* qfb   = (unsigned short*)alloc(6400ull * 512 * 2);
  unsigned short* kfb   = (unsigned short*)alloc(6400ull * 512 * 2);
  unsigned short* vfb   = (unsigned short*)alloc(6400ull * 512 * 2);
  unsigned short* attnO = (unsigned short*)alloc(6400ull * 512 * 2);
  float*          Gbuf  = (float*)alloc(64ull * 100 * 100 * 4);
  (void)ws_size; (void)in_sizes; (void)n_in; (void)out_size;

  k_f32_to_bf16<<<(6400 * 256 + 255) / 256, 256, 0, stream>>>(x, xbf, 6400 * 256);
  k_pack_w<<<(256 * 512 + 255) / 256, 256, 0, stream>>>(W0, W0p, 256, 512);
  k_pack_w<<<(512 * 512 + 255) / 256, 256, 0, stream>>>(Wq, Wqp, 512, 512);
  k_pack_w<<<(512 * 512 + 255) / 256, 256, 0, stream>>>(Wk, Wkp, 512, 512);
  k_pack_w<<<(512 * 512 + 255) / 256, 256, 0, stream>>>(Wv, Wvp, 512, 512);
  k_pack_w<<<(512 * 256 + 255) / 256, 256, 0, stream>>>(W1, W1p, 512, 256);

  dim3 g1(100, 8);  // M/64 x N/64, N=512
  k_gemm<<<g1, 32, 0, stream>>>(xbf, W0p, b0, nullptr, h, 256, 512, 1);
  k_gemm<<<g1, 32, 0, stream>>>(h, Wqp, nullptr, nullptr, qfb, 512, 512, 0);
  k_gemm<<<g1, 32, 0, stream>>>(h, Wkp, nullptr, nullptr, kfb, 512, 512, 0);
  k_gemm<<<g1, 32, 0, stream>>>(h, Wvp, nullptr, nullptr, vfb, 512, 512, 0);

  dim3 gs(64, 7, 5);  // batch x i-tiles x band j-tiles
  k_scores<<<gs, 32, 0, stream>>>(qfb, kfb, Gbuf);

  k_atten<<<6400, 256, 0, stream>>>(Gbuf, vfb, alphaOut, attnO);

  dim3 g2(100, 4);  // N=256
  k_gemm<<<g2, 32, 0, stream>>>(attnO, W1p, b1, y, nullptr, 512, 256, 1);
}
